// Cross_attention_76149770158366
// MI455X (gfx1250) — matrix operand order repaired and checked by reading, not compiled, not run on hardware
//
#include <hip/hip_runtime.h>

typedef _Float16 v16h __attribute__((ext_vector_type(16)));
typedef float    v8f  __attribute__((ext_vector_type(8)));

#define DIM      256
#define INNER    256
#define HEADS    8
#define DH       32
#define M_CTX    32
#define TOKENS   (2 * 8192)
#define TPB_TOK  16
#define NTHREADS 256
#define KT_W     8                      // K tiles (256 / 32)
#define ATTN_SCALE 0.17677669529663687f // 32^-0.5

// ---------------------------------------------------------------------------
// WMMA helpers
// ---------------------------------------------------------------------------
__device__ __forceinline__ v8f wmma_f16(v16h a, v16h b, v8f c) {
  // v_wmma_f32_16x16x32_f16: (neg_a, A, neg_b, B, c_mod, C, reuse_a, reuse_b)
  return __builtin_amdgcn_wmma_f32_16x16x32_f16(false, a, false, b, (short)0, c,
                                                false, false);
}

// Build a 16x32 f16 A fragment from row-major f32 data (global or LDS).
// Layout used here (contiguous K, not ISA table order): lane L holds row M=L%16;
//   lanes 0-15:  K = 0..15; lanes 16-31: K = 16..31.
__device__ __forceinline__ v16h a_frag_f32(const float* base, int stride,
                                           int row0, int k0, int lane) {
  const int r  = row0 + (lane & 15);
  const int kb = k0 + ((lane >> 4) & 1) * 16;
  const float4* p = (const float4*)(base + (size_t)r * stride + kb);
  float4 f0 = p[0];  // K kb+0..3
  float4 f1 = p[1];  // K kb+4..7
  float4 f2 = p[2];  // K kb+8..11
  float4 f3 = p[3];  // K kb+12..15
  v16h a;
  a[0]=(_Float16)f0.x;  a[1]=(_Float16)f0.y;  a[2]=(_Float16)f0.z;  a[3]=(_Float16)f0.w;
  a[4]=(_Float16)f1.x;  a[5]=(_Float16)f1.y;  a[6]=(_Float16)f1.z;  a[7]=(_Float16)f1.w;
  a[8]=(_Float16)f2.x;  a[9]=(_Float16)f2.y;  a[10]=(_Float16)f2.z; a[11]=(_Float16)f2.w;
  a[12]=(_Float16)f3.x; a[13]=(_Float16)f3.y; a[14]=(_Float16)f3.z; a[15]=(_Float16)f3.w;
  return a;
}

// Pre-packed B fragment: 16 halves per lane, contiguous (two b128 loads).
__device__ __forceinline__ v16h b_frag(const _Float16* __restrict__ P,
                                       int kt, int nt, int lane) {
  return *(const v16h*)(P + (((size_t)nt * KT_W + kt) * 32 + lane) * 16);
}

// ---------------------------------------------------------------------------
// Weight packing: f32 [Ktot,Ntot] row-major -> f16 WMMA B-fragment-major.
// Packed order (contiguous K as in a_frag_f32): lane L holds col N=L%16; lanes 0-15 K=0..15,
// lanes 16-31 K=16..31, two consecutive K per VGPR.
// ---------------------------------------------------------------------------
__global__ void pack_w(const float* __restrict__ W, _Float16* __restrict__ P,
                       int Ktot, int Ntot) {
  int i = blockIdx.x * NTHREADS + threadIdx.x;
  if (i >= Ktot * Ntot) return;
  int e    = i & 15;
  int lane = (i >> 4) & 31;
  int tile = i >> 9;               // nt * (Ktot/32) + kt
  int KT   = Ktot >> 5;
  int kt   = tile % KT;
  int nt   = tile / KT;
  int n    = nt * 16 + (lane & 15);
  int k    = kt * 32 + ((lane >> 4) & 1) * 16 + e;
  P[i] = (_Float16)W[(size_t)k * Ntot + n];
}

// ---------------------------------------------------------------------------
// Fused: q = x@Wq ; per token kv = y@Wkv ; softmax(q k^T / sqrt(d)) v ;
// out = o @ Wout + bout.  One block = 16 tokens, 8 waves.
// ---------------------------------------------------------------------------
__global__ __launch_bounds__(NTHREADS, 1)
void fused_xattn(const float* __restrict__ x, const float* __restrict__ y,
                 const _Float16* __restrict__ pWq,
                 const _Float16* __restrict__ pWkv,
                 const _Float16* __restrict__ pWout,
                 const float* __restrict__ bout, float* __restrict__ out) {
  __shared__ float kvS[M_CTX][516];   // k cols 0..255, v cols 256..511 (+pad)
  __shared__ float qS[TPB_TOK][260];
  __shared__ float oS[TPB_TOK][260];
  __shared__ float attnS[HEADS][M_CTX];

  const int tid  = threadIdx.x;
  const int w    = tid >> 5;         // wave 0..7
  const int lane = tid & 31;
  const int l15  = lane & 15;
  const int rhi  = ((lane >> 4) & 1) * 8;   // C-fragment row-half offset
  const int bt0  = blockIdx.x * TPB_TOK;

  // ---- Phase A: Q tile [16,256] = x[bt0..bt0+15] @ Wq --------------------
  {
    const float* xb = x + (size_t)bt0 * DIM;
    v8f c0 = {}, c1 = {};
#pragma unroll
    for (int kt = 0; kt < KT_W; ++kt) {
      v16h a  = a_frag_f32(xb, DIM, 0, kt * 32, lane);
      v16h b0 = b_frag(pWq, kt, 2 * w + 0, lane);
      v16h b1 = b_frag(pWq, kt, 2 * w + 1, lane);
      c0 = wmma_f16(a, b0, c0);
      c1 = wmma_f16(a, b1, c1);
    }
#pragma unroll
    for (int r = 0; r < 8; ++r) {
      qS[r + rhi][(2 * w + 0) * 16 + l15] = c0[r];
      qS[r + rhi][(2 * w + 1) * 16 + l15] = c1[r];
    }
  }
  __syncthreads();

  const int mhalf = w & 1;           // which 16 rows of the 32-row kv tile
  const int ngrp  = w >> 1;          // which 128-col slice of the 512 cols
  const int m0    = mhalf * 16;

  // ---- Per-token: kv GEMM + attention ------------------------------------
  for (int t = 0; t < TPB_TOK; ++t) {
    const float* yTok = y + (size_t)(bt0 + t) * (M_CTX * DIM);
    if (t + 1 < TPB_TOK) {
      // stream next token's 32KB context into cache (global_prefetch_b8)
      __builtin_prefetch(yTok + M_CTX * DIM + tid * 32, 0, 1);
    }

    // kv[32,512] = yTok[32,256] @ Wkv[256,512]
    v8f acc[8] = {};
    for (int kt = 0; kt < KT_W; ++kt) {
      v16h a = a_frag_f32(yTok, DIM, m0, kt * 32, lane);
#pragma unroll
      for (int j = 0; j < 8; ++j) {
        v16h b = b_frag(pWkv, kt, ngrp * 8 + j, lane);
        acc[j] = wmma_f16(a, b, acc[j]);
      }
    }
#pragma unroll
    for (int j = 0; j < 8; ++j)
#pragma unroll
      for (int r = 0; r < 8; ++r)
        kvS[m0 + r + rhi][ngrp * 128 + j * 16 + l15] = acc[j][r];
    __syncthreads();

    // attention: wave = head h, lane = context index m
    {
      float s = 0.f;
#pragma unroll
      for (int d = 0; d < DH; ++d)
        s += qS[t][w * DH + d] * kvS[lane][w * DH + d];
      s *= ATTN_SCALE;
      float mx = s;
#pragma unroll
      for (int off = 16; off > 0; off >>= 1)
        mx = fmaxf(mx, __shfl_xor(mx, off, 32));
      float e = __expf(s - mx);
      float sum = e;
#pragma unroll
      for (int off = 16; off > 0; off >>= 1)
        sum += __shfl_xor(sum, off, 32);
      attnS[w][lane] = e / sum;      // read back by the same wave below

      // lane = head-dim d now: o[h,d] = sum_m attn[m] * v[m, h*32+d]
      float o = 0.f;
#pragma unroll
      for (int m = 0; m < M_CTX; ++m)
        o += attnS[w][m] * kvS[m][INNER + w * DH + lane];
      oS[t][w * DH + lane] = o;
    }
    __syncthreads();   // protect kvS before next token overwrites it
  }

  // ---- Phase C: out[16,256] = oS @ Wout + bout ---------------------------
  {
    v8f c0 = {}, c1 = {};
    for (int kt = 0; kt < KT_W; ++kt) {
      v16h a  = a_frag_f32(&oS[0][0], 260, 0, kt * 32, lane);
      v16h b0 = b_frag(pWout, kt, 2 * w + 0, lane);
      v16h b1 = b_frag(pWout, kt, 2 * w + 1, lane);
      c0 = wmma_f16(a, b0, c0);
      c1 = wmma_f16(a, b1, c1);
    }
#pragma unroll
    for (int r = 0; r < 8; ++r) {
      int row  = bt0 + r + rhi;
      int col0 = (2 * w + 0) * 16 + l15;
      int col1 = (2 * w + 1) * 16 + l15;
      out[(size_t)row * DIM + col0] = c0[r] + bout[col0];
      out[(size_t)row * DIM + col1] = c1[r] + bout[col1];
    }
  }
}

// ---------------------------------------------------------------------------
extern "C" void kernel_launch(void* const* d_in, const int* in_sizes, int n_in,
                              void* d_out, int out_size, void* d_ws,
                              size_t ws_size, hipStream_t stream) {
  (void)in_sizes; (void)n_in; (void)out_size; (void)ws_size;
  const float* x    = (const float*)d_in[0];
  const float* y    = (const float*)d_in[1];
  const float* Wq   = (const float*)d_in[2];
  const float* Wkv  = (const float*)d_in[3];
  const float* Wout = (const float*)d_in[4];
  const float* bout = (const float*)d_in[5];
  float* out = (float*)d_out;

  _Float16* pWq   = (_Float16*)d_ws;               // 256*256 halves (128KB)
  _Float16* pWkv  = pWq + 256 * 256;               // 256*512 halves (256KB)
  _Float16* pWout = pWkv + 256 * 512;              // 256*256 halves (128KB)

  pack_w<<<(256 * 256 + NTHREADS - 1) / NTHREADS, NTHREADS, 0, stream>>>(
      Wq, pWq, 256, 256);
  pack_w<<<(256 * 512 + NTHREADS - 1) / NTHREADS, NTHREADS, 0, stream>>>(
      Wkv, pWkv, 256, 512);
  pack_w<<<(256 * 256 + NTHREADS - 1) / NTHREADS, NTHREADS, 0, stream>>>(
      Wout, pWout, 256, 256);

  fused_xattn<<<TOKENS / TPB_TOK, NTHREADS, 0, stream>>>(
      x, y, pWq, pWkv, pWout, bout, out);
}
